// AdaptiveSAGE_51247549776541
// MI455X (gfx1250) — matrix operand order repaired and checked by reading, not compiled, not run on hardware
//
#include <hip/hip_runtime.h>
#include <hip/hip_bf16.h>
#include <math.h>

typedef __attribute__((ext_vector_type(16))) __bf16 v16bf;
typedef __attribute__((ext_vector_type(8)))  __bf16 v8bf;
typedef __attribute__((ext_vector_type(4)))  __bf16 v4bf;
typedef __attribute__((ext_vector_type(8)))  float  v8f;

#define NSUB 80000
#define EDGES 800000
#define HDIM 128
#define MTILES 5000   // NSUB / 16

// ---------------------------------------------------------------------------
// Pack W matrices into per-lane WMMA B-fragment order (bf16).
// pWcat layout: [nt(8)][kc(8)][lane(32)][j(16)]  (K=256: k<128 -> W_r, else W_l)
// pWin  layout: [nt(8)][kc(4)][lane(32)][j(16)]  (K=128, W_in)
// B fragment convention: col n = lane&15 ; K = kc*32 + (lane>=16 ? 8 : 0) + (j&7) + (j>=8 ? 16 : 0)
// ---------------------------------------------------------------------------
__global__ void prep_weights_kernel(const float* __restrict__ Wl,
                                    const float* __restrict__ Wr,
                                    const float* __restrict__ Win,
                                    __bf16* __restrict__ pWcat,
                                    __bf16* __restrict__ pWin) {
    int idx = blockIdx.x * 256 + threadIdx.x;
    if (idx < 32768) {
        int j    = idx & 15;
        int lane = (idx >> 4) & 31;
        int kc   = (idx >> 9) & 7;
        int nt   = idx >> 12;
        int n    = nt * 16 + (lane & 15);
        int kk   = kc * 32 + ((lane >> 4) ? 8 : 0) + (j & 7) + ((j >> 3) ? 16 : 0);
        float v  = (kk < 128) ? Wr[n * 128 + kk] : Wl[n * 128 + (kk - 128)];
        pWcat[idx] = (__bf16)v;
    } else if (idx < 49152) {
        int i2   = idx - 32768;
        int j    = i2 & 15;
        int lane = (i2 >> 4) & 31;
        int kc   = (i2 >> 9) & 3;
        int nt   = i2 >> 11;
        int n    = nt * 16 + (lane & 15);
        int kk   = kc * 32 + ((lane >> 4) ? 8 : 0) + (j & 7) + ((j >> 3) ? 16 : 0);
        pWin[i2] = (__bf16)Win[n * 128 + kk];
    }
}

// ---------------------------------------------------------------------------
// Degree helpers
// ---------------------------------------------------------------------------
__global__ void zero_f32_kernel(float* __restrict__ p, int n) {
    int i = blockIdx.x * 256 + threadIdx.x;
    if (i < n) p[i] = 0.0f;
}
__global__ void zero_f32x4_kernel(float4* __restrict__ p, int n4) {
    int i = blockIdx.x * 256 + threadIdx.x;
    if (i < n4) p[i] = make_float4(0.f, 0.f, 0.f, 0.f);
}
__global__ void count_deg_kernel(const int* __restrict__ dst,
                                 float* __restrict__ deg) {
    int e = blockIdx.x * 256 + threadIdx.x;
    if (e < EDGES) atomicAdd(&deg[dst[e]], 1.0f);
}
__global__ void invert_deg_kernel(float* __restrict__ deg) {
    int i = blockIdx.x * 256 + threadIdx.x;
    if (i < NSUB) deg[i] = 1.0f / fmaxf(deg[i], 1.0f);
}

// ---------------------------------------------------------------------------
// Input projection: H0[i,:] = x_full[subset[i],:] @ W_in.T + b_in  (bf16 out)
// One wave per 16-row M-tile, 8 N-tiles of 16, K=128 via 4 WMMA k-chunks.
// ---------------------------------------------------------------------------
__global__ void input_gemm_kernel(const float* __restrict__ x_full,
                                  const int* __restrict__ subset,
                                  const __bf16* __restrict__ pWin,
                                  const float* __restrict__ b_in,
                                  __bf16* __restrict__ H0) {
    __shared__ __bf16 lds[16384];            // 32 KB packed W_in fragments
    {
        const uint4* s4 = (const uint4*)pWin;
        uint4* d4 = (uint4*)lds;
        for (int i = threadIdx.x; i < 2048; i += 256) d4[i] = s4[i];
    }
    __syncthreads();

    int lane = threadIdx.x & 31;
    int wave = threadIdx.x >> 5;
    int tile = blockIdx.x * 8 + wave;
    if (tile >= MTILES) return;
    int m    = lane & 15;
    int half = lane >> 4;
    int row  = tile * 16 + m;
    const float* xr = x_full + (long)subset[row] * 128;

    v8f acc[8];
    for (int nt = 0; nt < 8; ++nt) {
        float bv = b_in[nt * 16 + m];
        for (int r = 0; r < 8; ++r) acc[nt][r] = bv;
    }

    for (int kc = 0; kc < 4; ++kc) {
        int k0 = kc * 32 + (half ? 8 : 0);
        int k1 = kc * 32 + (half ? 24 : 16);
        float4 f0 = *(const float4*)(xr + k0);
        float4 f1 = *(const float4*)(xr + k0 + 4);
        float4 f2 = *(const float4*)(xr + k1);
        float4 f3 = *(const float4*)(xr + k1 + 4);
        v16bf a;
        a[0]=(__bf16)f0.x; a[1]=(__bf16)f0.y; a[2]=(__bf16)f0.z; a[3]=(__bf16)f0.w;
        a[4]=(__bf16)f1.x; a[5]=(__bf16)f1.y; a[6]=(__bf16)f1.z; a[7]=(__bf16)f1.w;
        a[8]=(__bf16)f2.x; a[9]=(__bf16)f2.y; a[10]=(__bf16)f2.z; a[11]=(__bf16)f2.w;
        a[12]=(__bf16)f3.x; a[13]=(__bf16)f3.y; a[14]=(__bf16)f3.z; a[15]=(__bf16)f3.w;
        for (int nt = 0; nt < 8; ++nt) {
            const __bf16* bp = lds + ((nt * 4 + kc) * 32 + lane) * 16;
            v8bf b0 = *(const v8bf*)bp;
            v8bf b1 = *(const v8bf*)(bp + 8);
            v16bf b;
            for (int i = 0; i < 8; ++i) { b[i] = b0[i]; b[8 + i] = b1[i]; }
            acc[nt] = __builtin_amdgcn_wmma_f32_16x16x32_bf16(
                false, a, false, b, (short)0, acc[nt], false, false);
        }
    }

    __bf16* out = H0 + (long)tile * 16 * 128;
    for (int nt = 0; nt < 8; ++nt)
        for (int r = 0; r < 8; ++r) {
            int mm = half * 8 + r;
            out[mm * 128 + nt * 16 + m] = (__bf16)acc[nt][r];
        }
}

// ---------------------------------------------------------------------------
// Edge scatter-add: agg[dst,:] += H[src,:]   (one wave per edge, 4 feats/lane)
// ---------------------------------------------------------------------------
__global__ void scatter_kernel(const __bf16* __restrict__ H,
                               const int* __restrict__ src,
                               const int* __restrict__ dst,
                               float* __restrict__ agg) {
    int e = blockIdx.x * 8 + (threadIdx.x >> 5);
    if (e >= EDGES) return;
    int lane = threadIdx.x & 31;
    int s = src[e], d = dst[e];
    v4bf h = *(const v4bf*)(H + (long)s * 128 + lane * 4);
    float* ap = agg + (long)d * 128 + lane * 4;
    atomicAdd(ap + 0, (float)h[0]);
    atomicAdd(ap + 1, (float)h[1]);
    atomicAdd(ap + 2, (float)h[2]);
    atomicAdd(ap + 3, (float)h[3]);
}

// ---------------------------------------------------------------------------
// SAGE step GEMM: Hout = relu([Hin | agg*invdeg] @ [W_r|W_l].T + b_l)  K=256
// ---------------------------------------------------------------------------
__global__ void step_gemm_kernel(const __bf16* __restrict__ Hin,
                                 const float* __restrict__ agg,
                                 const float* __restrict__ invdeg,
                                 const __bf16* __restrict__ pWcat,
                                 const float* __restrict__ b_l,
                                 __bf16* __restrict__ Hout) {
    __shared__ __bf16 lds[32768];            // 64 KB packed [W_r|W_l] fragments
    {
        const uint4* s4 = (const uint4*)pWcat;
        uint4* d4 = (uint4*)lds;
        for (int i = threadIdx.x; i < 4096; i += 256) d4[i] = s4[i];
    }
    __syncthreads();

    int lane = threadIdx.x & 31;
    int wave = threadIdx.x >> 5;
    int tile = blockIdx.x * 8 + wave;
    if (tile >= MTILES) return;
    int m    = lane & 15;
    int half = lane >> 4;
    int row  = tile * 16 + m;
    const __bf16* hr = Hin + (long)row * 128;
    const float*  ar = agg + (long)row * 128;
    float idg = invdeg[row];

    v8f acc[8];
    for (int nt = 0; nt < 8; ++nt) {
        float bv = b_l[nt * 16 + m];
        for (int r = 0; r < 8; ++r) acc[nt][r] = bv;
    }

    for (int kc = 0; kc < 8; ++kc) {
        v16bf a;
        if (kc < 4) {
            int k0 = kc * 32 + (half ? 8 : 0);
            int k1 = kc * 32 + (half ? 24 : 16);
            v8bf h0 = *(const v8bf*)(hr + k0);
            v8bf h1 = *(const v8bf*)(hr + k1);
            for (int i = 0; i < 8; ++i) { a[i] = h0[i]; a[8 + i] = h1[i]; }
        } else {
            int k0 = (kc - 4) * 32 + (half ? 8 : 0);
            int k1 = (kc - 4) * 32 + (half ? 24 : 16);
            float4 f0 = *(const float4*)(ar + k0);
            float4 f1 = *(const float4*)(ar + k0 + 4);
            float4 f2 = *(const float4*)(ar + k1);
            float4 f3 = *(const float4*)(ar + k1 + 4);
            a[0]=(__bf16)(f0.x*idg); a[1]=(__bf16)(f0.y*idg);
            a[2]=(__bf16)(f0.z*idg); a[3]=(__bf16)(f0.w*idg);
            a[4]=(__bf16)(f1.x*idg); a[5]=(__bf16)(f1.y*idg);
            a[6]=(__bf16)(f1.z*idg); a[7]=(__bf16)(f1.w*idg);
            a[8]=(__bf16)(f2.x*idg); a[9]=(__bf16)(f2.y*idg);
            a[10]=(__bf16)(f2.z*idg); a[11]=(__bf16)(f2.w*idg);
            a[12]=(__bf16)(f3.x*idg); a[13]=(__bf16)(f3.y*idg);
            a[14]=(__bf16)(f3.z*idg); a[15]=(__bf16)(f3.w*idg);
        }
        for (int nt = 0; nt < 8; ++nt) {
            const __bf16* bp = lds + ((nt * 8 + kc) * 32 + lane) * 16;
            v8bf b0 = *(const v8bf*)bp;
            v8bf b1 = *(const v8bf*)(bp + 8);
            v16bf b;
            for (int i = 0; i < 8; ++i) { b[i] = b0[i]; b[8 + i] = b1[i]; }
            acc[nt] = __builtin_amdgcn_wmma_f32_16x16x32_bf16(
                false, a, false, b, (short)0, acc[nt], false, false);
        }
    }

    __bf16* out = Hout + (long)tile * 16 * 128;
    for (int nt = 0; nt < 8; ++nt)
        for (int r = 0; r < 8; ++r) {
            int mm = half * 8 + r;
            out[mm * 128 + nt * 16 + m] = (__bf16)fmaxf(acc[nt][r], 0.0f);
        }
}

// ---------------------------------------------------------------------------
// Stash rows 0 and 1 (h_u, h_v) of the current H as f32 for the ACT head.
// ---------------------------------------------------------------------------
__global__ void stash_kernel(const __bf16* __restrict__ H,
                             float* __restrict__ stash_k) {
    int t = threadIdx.x;             // 256 threads: row0 || row1
    stash_k[t] = (float)H[t];
}

// ---------------------------------------------------------------------------
// ACT head: per-depth edge-score MLP + halting MLP, alpha normalization.
// Single block, 128 threads; serial scalar state on thread 0.
// ---------------------------------------------------------------------------
__global__ void head_kernel(const float* __restrict__ stash,
                            const float* __restrict__ We1, const float* __restrict__ be1,
                            const float* __restrict__ We2, const float* __restrict__ be2,
                            const float* __restrict__ Wh1, const float* __restrict__ bh1,
                            const float* __restrict__ Wh2, const float* __restrict__ bh2,
                            float* __restrict__ out) {
    __shared__ float shu[128], shv[128], red[128];
    __shared__ float s_score;
    int t = threadIdx.x;
    float alpha[5], sc[5];
    float p_not = 1.0f;

    for (int k = 0; k < 5; ++k) {
        shu[t] = stash[k * 256 + t];
        shv[t] = stash[k * 256 + 128 + t];
        __syncthreads();

        // e1 = relu(W_e1 @ [hu, hv, hu*hv] + b_e1), thread t -> output t
        const float* w = We1 + t * 384;
        float av = be1[t];
        for (int j = 0; j < 128; ++j) av += w[j] * shu[j];
        for (int j = 0; j < 128; ++j) av += w[128 + j] * shv[j];
        for (int j = 0; j < 128; ++j) av += w[256 + j] * (shu[j] * shv[j]);
        av = fmaxf(av, 0.0f);
        red[t] = We2[t] * av;
        __syncthreads();
        if (t == 0) {
            float s = be2[0];
            for (int j = 0; j < 128; ++j) s += red[j];
            s_score = s;
        }
        __syncthreads();
        float score = s_score;

        // h1 = relu(W_h1 @ [hu, hv, score] + b_h1), threads 0..63
        float hv1 = 0.0f;
        if (t < 64) {
            const float* w1 = Wh1 + t * 257;
            float a2 = bh1[t];
            for (int j = 0; j < 128; ++j) a2 += w1[j] * shu[j];
            for (int j = 0; j < 128; ++j) a2 += w1[128 + j] * shv[j];
            a2 += w1[256] * score;
            hv1 = fmaxf(a2, 0.0f) * Wh2[t];
        }
        red[t] = (t < 64) ? hv1 : 0.0f;
        __syncthreads();
        if (t == 0) {
            float z = bh2[0];
            for (int j = 0; j < 64; ++j) z += red[j];
            float p = 1.0f / (1.0f + expf(-z));
            alpha[k] = p * p_not;
            sc[k]    = score;
            p_not   *= (1.0f - p);
        }
        __syncthreads();
    }

    if (t == 0) {
        float s = 0.0f;
        for (int k = 0; k < 5; ++k) s += alpha[k];
        float inv = 1.0f / (s + 1e-8f);
        float fs = 0.0f, ed = 0.0f;
        for (int k = 0; k < 5; ++k) {
            float a = alpha[k] * inv;
            out[2 + k] = a;
            fs += a * sc[k];
            ed += a * (float)(k + 1);
        }
        out[0] = fs;
        out[1] = ed;
    }
}

// ---------------------------------------------------------------------------
extern "C" void kernel_launch(void* const* d_in, const int* in_sizes, int n_in,
                              void* d_out, int out_size, void* d_ws, size_t ws_size,
                              hipStream_t stream) {
    const float* x_full = (const float*)d_in[0];
    const float* W_in   = (const float*)d_in[1];
    const float* b_in   = (const float*)d_in[2];
    const float* W_l    = (const float*)d_in[3];
    const float* b_l    = (const float*)d_in[4];
    const float* W_r    = (const float*)d_in[5];
    const float* W_e1   = (const float*)d_in[6];
    const float* b_e1   = (const float*)d_in[7];
    const float* W_e2   = (const float*)d_in[8];
    const float* b_e2   = (const float*)d_in[9];
    const float* W_h1   = (const float*)d_in[10];
    const float* b_h1   = (const float*)d_in[11];
    const float* W_h2   = (const float*)d_in[12];
    const float* b_h2   = (const float*)d_in[13];
    const int*   subset = (const int*)d_in[14];
    const int*   edge   = (const int*)d_in[15];
    const int*   e_src  = edge;
    const int*   e_dst  = edge + EDGES;

    char* ws = (char*)d_ws;
    __bf16* Hb0    = (__bf16*)(ws);                 // 20,480,000 B
    __bf16* Hb1    = (__bf16*)(ws + 20480000);      // 20,480,000 B
    float*  agg    = (float*) (ws + 40960000);      // 40,960,000 B
    float*  invdeg = (float*) (ws + 81920000);      //    320,000 B
    __bf16* pWcat  = (__bf16*)(ws + 82240000);      //     65,536 B
    __bf16* pWin   = (__bf16*)(ws + 82305536);      //     32,768 B
    float*  stash  = (float*) (ws + 82338304);      //      5,120 B

    // Weight packing + degree (once per call)
    prep_weights_kernel<<<192, 256, 0, stream>>>(W_l, W_r, W_in, pWcat, pWin);
    zero_f32_kernel<<<(NSUB + 255) / 256, 256, 0, stream>>>(invdeg, NSUB);
    count_deg_kernel<<<(EDGES + 255) / 256, 256, 0, stream>>>(e_dst, invdeg);
    invert_deg_kernel<<<(NSUB + 255) / 256, 256, 0, stream>>>(invdeg);

    // H0 = gather + input projection (WMMA)
    input_gemm_kernel<<<625, 256, 0, stream>>>(x_full, subset, pWin, b_in, Hb0);

    __bf16* Hin = Hb0;
    __bf16* Hout = Hb1;
    for (int k = 0; k < 5; ++k) {
        zero_f32x4_kernel<<<10000, 256, 0, stream>>>((float4*)agg, NSUB * 32);
        scatter_kernel<<<100000, 256, 0, stream>>>(Hin, e_src, e_dst, agg);
        step_gemm_kernel<<<625, 256, 0, stream>>>(Hin, agg, invdeg, pWcat, b_l, Hout);
        stash_kernel<<<1, 256, 0, stream>>>(Hout, stash + k * 256);
        __bf16* tmp = Hin; Hin = Hout; Hout = tmp;
    }

    head_kernel<<<1, 128, 0, stream>>>(stash, W_e1, b_e1, W_e2, b_e2,
                                       W_h1, b_h1, W_h2, b_h2, (float*)d_out);
}
